// Xerxes2MoeMLPStack_8856222564599
// MI455X (gfx1250) — matrix operand order repaired and checked by reading, not compiled, not run on hardware
//
#include <hip/hip_runtime.h>
#include <hip/hip_bf16.h>
#include <stdint.h>

// Problem constants (fixed by the reference)
#define T_TOK 16384
#define HID   2048
#define INTER 5632
#define NEXP  8

// Tiling
#define BM 128
#define BN 64
#define BK 64
#define LDA 72          // BK + 8 pad (144B row stride, 16B aligned)
#define LDB 72
#define TPE (T_TOK / BM)   // max M-tiles per expert = 128
#define NTHREADS 256       // 8 waves (wave32)

typedef __attribute__((ext_vector_type(16))) __bf16 v16bf;
typedef __attribute__((ext_vector_type(8)))  float  v8f;

// Vector/pointer types matching the async-LDS builtin signature
// (param type per hipcc diagnostic: "int __vector(4) __device__ *")
typedef int v4i __attribute__((vector_size(16)));
typedef __attribute__((address_space(1))) v4i* gv4i_p;   // global
typedef __attribute__((address_space(3))) v4i* lv4i_p;   // LDS

#ifndef HAVE_ASYNC_LDS
#if defined(__has_builtin)
#if __has_builtin(__builtin_amdgcn_global_load_async_to_lds_b128) && \
    __has_builtin(__builtin_amdgcn_s_wait_asynccnt)
#define HAVE_ASYNC_LDS 1
#endif
#endif
#endif
#ifndef HAVE_ASYNC_LDS
#define HAVE_ASYNC_LDS 0
#endif

__device__ __forceinline__ int imin(int a, int b) { return a < b ? a : b; }

// Native bf16 conversion (RNE) — lets the compiler pick v_cvt hardware path
__device__ __forceinline__ unsigned short f32_to_bf16(float f) {
  union { __bf16 b; unsigned short u; } cv;
  cv.b = (__bf16)f;
  return cv.u;
}
__device__ __forceinline__ unsigned pack2_bf16(float a, float b) {
  return (unsigned)f32_to_bf16(a) | ((unsigned)f32_to_bf16(b) << 16);
}

// A fragment, 16x32 bf16: lanes 0-15 row M=lane hold K=0..15;
// lanes 16-31 row M=lane-16 hold K=16..31. LDS A is [m][k], row stride LDA.
__device__ __forceinline__ v16bf ld_frag_a(const unsigned short* As, int m0,
                                           int ks, int lane) {
  const int half = lane >> 4;
  const int row  = m0 + (lane & 15);
  const unsigned short* p = As + row * LDA + ks * 32 + half * 16;
  union { uint4 q[2]; v16bf v; } u;
  u.q[0] = *(const uint4*)(p);
  u.q[1] = *(const uint4*)(p + 8);
  return u.v;
}

// B fragment, 32x16 bf16: lane n=lane&15 is column N; lanes 0-15 hold K=0..15,
// lanes 16-31 hold K=16..31 (VGPR j packs K=2j,2j+1). LDS B stored transposed
// [n][k] so each lane reads 16 contiguous bf16.
__device__ __forceinline__ v16bf ld_frag_b(const unsigned short* Bs, int n0,
                                           int ks, int lane) {
  const int half = lane >> 4;
  const int n    = n0 + (lane & 15);
  const unsigned short* p = Bs + n * LDB + ks * 32 + half * 16;
  union { uint4 q[2]; v16bf v; } u;
  u.q[0] = *(const uint4*)(p);
  u.q[1] = *(const uint4*)(p + 8);
  return u.v;
}

// ---------------- Kernel 1: gate & up GEMMs + SwiGLU -> fused bf16 -----------
__global__ __launch_bounds__(NTHREADS)
void moe_gateup_kernel(const float* __restrict__ x,
                       const int*   __restrict__ gs,
                       const float* __restrict__ wg,
                       const float* __restrict__ wu,
                       unsigned short* __restrict__ fused) {
  const int e  = blockIdx.y / TPE;
  const int mt = blockIdx.y % TPE;
  int off = 0;
#pragma unroll
  for (int i = 0; i < NEXP; ++i) off += (i < e) ? gs[i] : 0;
  const int sz = gs[e];
  if (mt * BM >= sz) return;
  const int row0 = off + mt * BM;
  const int n0   = blockIdx.x * BN;
  const int mlim = imin(BM, sz - mt * BM) - 1;   // last valid in-tile row

  __shared__ __align__(16) unsigned short As[BM * LDA];
  __shared__ __align__(16) unsigned short Bg[BN * LDB];
  __shared__ __align__(16) unsigned short Bu[BN * LDB];

  const int tid  = threadIdx.x;
  const int lane = tid & 31;
  const int wid  = tid >> 5;
  const int wm   = wid >> 1;      // 0..3 : 4 waves along M (4*32 = 128)
  const int wn   = wid & 1;       // 0..1 : 2 waves along N (2*32 = 64)
  const int half = lane >> 4;
  const int l16  = lane & 15;

  const float* wge = wg + (size_t)e * HID * INTER;
  const float* wue = wu + (size_t)e * HID * INTER;

  v8f cg[2][2], cu[2][2];
  const v8f vzero = {0.f, 0.f, 0.f, 0.f, 0.f, 0.f, 0.f, 0.f};
#pragma unroll
  for (int i = 0; i < 2; ++i)
#pragma unroll
    for (int j = 0; j < 2; ++j) { cg[i][j] = vzero; cu[i][j] = vzero; }

  for (int k0 = 0; k0 < HID; k0 += BK) {
    // A tile: 128x64 f32 -> bf16 into LDS [m][k]. Unconditional loads with a
    // clamped row index (padding rows hold duplicate data, masked on store).
#pragma unroll
    for (int it = 0; it < 8; ++it) {
      int idx = tid + it * NTHREADS;          // 0..2047 float4s
      int m   = idx >> 4;                     // 16 float4 per row
      int c4  = idx & 15;
      int mc  = imin(m, mlim);
      float4 v = *(const float4*)(x + (size_t)(row0 + mc) * HID + k0 + c4 * 4);
      *(uint2*)(&As[m * LDA + c4 * 4]) =
          make_uint2(pack2_bf16(v.x, v.y), pack2_bf16(v.z, v.w));
    }
    // B tiles (gate & up): 64x64 f32 -> bf16, stored transposed [n][k].
    // Process k in pairs so each LDS store is a packed b32.
#pragma unroll
    for (int it = 0; it < 2; ++it) {
      int idx = tid + it * NTHREADS;          // 0..511 (k-pair, n4) tasks
      int k   = (idx >> 4) * 2;
      int n4  = (idx & 15) * 4;
      size_t goff = (size_t)(k0 + k) * INTER + n0 + n4;
      float4 g0 = *(const float4*)(wge + goff);
      float4 g1 = *(const float4*)(wge + goff + INTER);
      float4 u0 = *(const float4*)(wue + goff);
      float4 u1 = *(const float4*)(wue + goff + INTER);
      *(unsigned*)(&Bg[(n4 + 0) * LDB + k]) = pack2_bf16(g0.x, g1.x);
      *(unsigned*)(&Bg[(n4 + 1) * LDB + k]) = pack2_bf16(g0.y, g1.y);
      *(unsigned*)(&Bg[(n4 + 2) * LDB + k]) = pack2_bf16(g0.z, g1.z);
      *(unsigned*)(&Bg[(n4 + 3) * LDB + k]) = pack2_bf16(g0.w, g1.w);
      *(unsigned*)(&Bu[(n4 + 0) * LDB + k]) = pack2_bf16(u0.x, u1.x);
      *(unsigned*)(&Bu[(n4 + 1) * LDB + k]) = pack2_bf16(u0.y, u1.y);
      *(unsigned*)(&Bu[(n4 + 2) * LDB + k]) = pack2_bf16(u0.z, u1.z);
      *(unsigned*)(&Bu[(n4 + 3) * LDB + k]) = pack2_bf16(u0.w, u1.w);
    }
    __syncthreads();
#pragma unroll
    for (int ks = 0; ks < 2; ++ks) {
      v16bf a[2], bg2[2], bu2[2];
#pragma unroll
      for (int i = 0; i < 2; ++i) a[i] = ld_frag_a(As, wm * 32 + i * 16, ks, lane);
#pragma unroll
      for (int j = 0; j < 2; ++j) {
        bg2[j] = ld_frag_b(Bg, wn * 32 + j * 16, ks, lane);
        bu2[j] = ld_frag_b(Bu, wn * 32 + j * 16, ks, lane);
      }
#pragma unroll
      for (int i = 0; i < 2; ++i)
#pragma unroll
        for (int j = 0; j < 2; ++j) {
          cg[i][j] = __builtin_amdgcn_wmma_f32_16x16x32_bf16(
              false, a[i], false, bg2[j], (short)0, cg[i][j], false, false);
          cu[i][j] = __builtin_amdgcn_wmma_f32_16x16x32_bf16(
              false, a[i], false, bu2[j], (short)0, cu[i][j], false, false);
        }
    }
    __syncthreads();
  }

  // Epilogue: fused = silu(gate) * up, bf16 store. C layout: VGPR r -> M=r+8*half, N=l16.
#pragma unroll
  for (int i = 0; i < 2; ++i)
#pragma unroll
    for (int j = 0; j < 2; ++j)
#pragma unroll
      for (int r = 0; r < 8; ++r) {
        int m = wm * 32 + i * 16 + r + half * 8;
        if (mt * BM + m < sz) {
          float g = cg[i][j][r];
          float u = cu[i][j][r];
          float f = g / (1.f + __expf(-g)) * u;
          int col = n0 + wn * 32 + j * 16 + l16;
          fused[(size_t)(row0 + m) * INTER + col] = f32_to_bf16(f);
        }
      }
}

// ---------------- Kernel 2: down GEMM (fused bf16 x w_down) -> f32 out -------
__global__ __launch_bounds__(NTHREADS)
void moe_down_kernel(const unsigned short* __restrict__ fused,
                     const int*   __restrict__ gs,
                     const float* __restrict__ wd,
                     float* __restrict__ out) {
  const int e  = blockIdx.y / TPE;
  const int mt = blockIdx.y % TPE;
  int off = 0;
#pragma unroll
  for (int i = 0; i < NEXP; ++i) off += (i < e) ? gs[i] : 0;
  const int sz = gs[e];
  if (mt * BM >= sz) return;
  const int row0 = off + mt * BM;
  const int n0   = blockIdx.x * BN;
  const int mlim = imin(BM, sz - mt * BM) - 1;

  __shared__ __align__(16) unsigned short As[BM * LDA];
  __shared__ __align__(16) unsigned short Bs[BN * LDB];

  const int tid  = threadIdx.x;
  const int lane = tid & 31;
  const int wid  = tid >> 5;
  const int wm   = wid >> 1;
  const int wn   = wid & 1;
  const int half = lane >> 4;
  const int l16  = lane & 15;

  const float* wde = wd + (size_t)e * INTER * HID;

  v8f c[2][2];
  const v8f vzero = {0.f, 0.f, 0.f, 0.f, 0.f, 0.f, 0.f, 0.f};
#pragma unroll
  for (int i = 0; i < 2; ++i)
#pragma unroll
    for (int j = 0; j < 2; ++j) c[i][j] = vzero;

  for (int k0 = 0; k0 < INTER; k0 += BK) {
    // A tile: 128x64 bf16 straight copy -> CDNA5 async global->LDS DMA path
#pragma unroll
    for (int it = 0; it < 4; ++it) {
      int idx = tid + it * NTHREADS;          // 0..1023 uint4 (8 bf16 each)
      int m   = idx >> 3;                     // 8 uint4 per row
      int c8  = idx & 7;
      int mc  = imin(m, mlim);
      const unsigned short* gp = fused + (size_t)(row0 + mc) * INTER + k0 + c8 * 8;
      unsigned short* lp = &As[m * LDA + c8 * 8];
#if HAVE_ASYNC_LDS
      __builtin_amdgcn_global_load_async_to_lds_b128((gv4i_p)gp, (lv4i_p)lp, 0, 0);
#else
      *(uint4*)lp = *(const uint4*)gp;
#endif
    }
    // B tile: 64x64 f32 -> bf16, transposed [n][k], packed b32 LDS stores
#pragma unroll
    for (int it = 0; it < 2; ++it) {
      int idx = tid + it * NTHREADS;
      int k   = (idx >> 4) * 2;
      int n4  = (idx & 15) * 4;
      size_t goff = (size_t)(k0 + k) * HID + n0 + n4;
      float4 d0 = *(const float4*)(wde + goff);
      float4 d1 = *(const float4*)(wde + goff + HID);
      *(unsigned*)(&Bs[(n4 + 0) * LDB + k]) = pack2_bf16(d0.x, d1.x);
      *(unsigned*)(&Bs[(n4 + 1) * LDB + k]) = pack2_bf16(d0.y, d1.y);
      *(unsigned*)(&Bs[(n4 + 2) * LDB + k]) = pack2_bf16(d0.z, d1.z);
      *(unsigned*)(&Bs[(n4 + 3) * LDB + k]) = pack2_bf16(d0.w, d1.w);
    }
#if HAVE_ASYNC_LDS
    __builtin_amdgcn_s_wait_asynccnt(0);
#endif
    __syncthreads();
#pragma unroll
    for (int ks = 0; ks < 2; ++ks) {
      v16bf a[2], b2[2];
#pragma unroll
      for (int i = 0; i < 2; ++i) a[i] = ld_frag_a(As, wm * 32 + i * 16, ks, lane);
#pragma unroll
      for (int j = 0; j < 2; ++j) b2[j] = ld_frag_b(Bs, wn * 32 + j * 16, ks, lane);
#pragma unroll
      for (int i = 0; i < 2; ++i)
#pragma unroll
        for (int j = 0; j < 2; ++j)
          c[i][j] = __builtin_amdgcn_wmma_f32_16x16x32_bf16(
              false, a[i], false, b2[j], (short)0, c[i][j], false, false);
    }
    __syncthreads();
  }

#pragma unroll
  for (int i = 0; i < 2; ++i)
#pragma unroll
    for (int j = 0; j < 2; ++j)
#pragma unroll
      for (int r = 0; r < 8; ++r) {
        int m = wm * 32 + i * 16 + r + half * 8;
        if (mt * BM + m < sz) {
          int col = n0 + wn * 32 + j * 16 + l16;
          out[(size_t)(row0 + m) * HID + col] = c[i][j][r];
        }
      }
}

extern "C" void kernel_launch(void* const* d_in, const int* in_sizes, int n_in,
                              void* d_out, int out_size, void* d_ws, size_t ws_size,
                              hipStream_t stream) {
  (void)in_sizes; (void)n_in; (void)out_size; (void)ws_size;
  const float* x  = (const float*)d_in[0];
  const int*   gs = (const int*)d_in[1];
  const float* wg = (const float*)d_in[2];
  const float* wu = (const float*)d_in[3];
  const float* wd = (const float*)d_in[4];
  float* out = (float*)d_out;
  unsigned short* fused = (unsigned short*)d_ws;   // T_TOK*INTER bf16 = 184 MB

  dim3 grid1(INTER / BN, NEXP * TPE);
  moe_gateup_kernel<<<grid1, NTHREADS, 0, stream>>>(x, gs, wg, wu, fused);

  dim3 grid2(HID / BN, NEXP * TPE);
  moe_down_kernel<<<grid2, NTHREADS, 0, stream>>>(fused, gs, wd, out);
}
